// SelfAttention_37769942401114
// MI455X (gfx1250) — matrix operand order repaired and checked by reading, not compiled, not run on hardware
//
#include <hip/hip_runtime.h>
#include <hip/hip_bf16.h>

typedef __attribute__((ext_vector_type(16))) _Float16 v16h;
typedef __attribute__((ext_vector_type(8)))  float    v8f;
typedef __attribute__((ext_vector_type(4)))  uint32_t u32x4;
typedef __attribute__((ext_vector_type(8)))  uint32_t u32x8;

#define HEADS 8
#define HEAD_DIM 64
#define HIDDEN 512            // HEADS*HEAD_DIM
#define CCH 512               // channels C
#define NPIX 1024             // H*W
#define NB 16                 // batch
#define QKV_M 1536            // 3*HIDDEN
#define ATTN_SCALE 0.125f     // HEAD_DIM^-0.5 ; also the (faithful) RMSNorm eps

// ---------------- WMMA helpers (gfx1250 wave32 layouts) ----------------

// D = A(16x32 f16) * B(32x16 f16) + C(16x16 f32)
__device__ __forceinline__ v8f wmma16(v16h a, v16h b, v8f c) {
  return __builtin_amdgcn_wmma_f32_16x16x32_f16(
      /*neg_a=*/false, a, /*neg_b=*/false, b,
      /*c_mod=*/(short)0, c, /*reuse_a=*/false, /*reuse_b=*/false);
}

// A fragment: p points at (row, k0) in a row-major [M,K] f16 matrix with row
// stride `ldk` applied by the caller. Lane layout (16-bit A 16x32):
// lanes<16 hold K={0..7,16..23}, lanes>=16 hold K={8..15,24..31}.
__device__ __forceinline__ v16h frag_a(const _Float16* p, int lane_hi) {
  v16h f;
  uint32_t* d = reinterpret_cast<uint32_t*>(&f);
  const uint32_t* q0 = reinterpret_cast<const uint32_t*>(p + (lane_hi ? 8 : 0));
  const uint32_t* q1 = reinterpret_cast<const uint32_t*>(p + 16 + (lane_hi ? 8 : 0));
#pragma unroll
  for (int j = 0; j < 4; ++j) d[j] = q0[j];
#pragma unroll
  for (int j = 0; j < 4; ++j) d[4 + j] = q1[j];
  return f;
}

// B fragment: p points at (col, k0) in a row-major [N,K] f16 matrix
// (column-contiguous K). Lanes<16 hold K={0..7,16..23}, lanes>=16 hold K={8..15,24..31}.
__device__ __forceinline__ v16h frag_b(const _Float16* p, int lane_hi) {
  v16h f;
  uint32_t* d = reinterpret_cast<uint32_t*>(&f);
  const uint32_t* q0 = reinterpret_cast<const uint32_t*>(p + (lane_hi ? 8 : 0));
  const uint32_t* q1 = reinterpret_cast<const uint32_t*>(p + 16 + (lane_hi ? 8 : 0));
#pragma unroll
  for (int j = 0; j < 4; ++j) d[j] = q0[j];
#pragma unroll
  for (int j = 0; j < 4; ++j) d[4 + j] = q1[j];
  return f;
}

// ---------------- Tensor Data Mover: 2-D f16 tile -> LDS ----------------
// D# per CDNA5 ISA §8.3/§8.4 (2 SGPR groups; VADDR2/3 = NULL => up-to-2D tensor).
// tile_d0 elements along contiguous dim (K), tile_d1 rows; dims/strides in
// 2-byte elements; addresses in bytes.
__device__ __forceinline__ void tdm_load_2d_f16(uint32_t lds_off,
                                                const _Float16* gaddr,
                                                uint32_t tensor_d0, uint32_t tensor_d1,
                                                uint32_t tile_d0, uint32_t tile_d1,
                                                uint32_t stride0) {
  uint64_t ga = (uint64_t)(uintptr_t)gaddr;
  u32x4 g0;
  g0[0] = 1u;                                   // count=1, user descriptor
  g0[1] = lds_off;                              // lds_addr (bytes)
  g0[2] = (uint32_t)ga;                         // global_addr[31:0]
  g0[3] = (uint32_t)(ga >> 32) | (2u << 30);    // global_addr[56:32] | type=2
  u32x8 g1;
  g1[0] = 1u << 16;                             // data_size=1 (2 bytes)
  g1[1] = (tensor_d0 & 0xFFFFu) << 16;          // tensor_dim0[15:0]
  g1[2] = (tensor_d0 >> 16) | ((tensor_d1 & 0xFFFFu) << 16);
  g1[3] = (tensor_d1 >> 16) | (tile_d0 << 16);  // tensor_dim1[31:16] | tile_dim0
  g1[4] = tile_d1;                              // tile_dim1 (tile_dim2 = 0)
  g1[5] = stride0;                              // tensor_dim0_stride[31:0]
  g1[6] = 0;                                    // stride0[47:32] | dim1_stride lo
  g1[7] = 0;
  asm volatile("tensor_load_to_lds %0, %1" :: "s"(g0), "s"(g1) : "memory");
}

// ---------------- Stage 0: conversions ----------------

__global__ __launch_bounds__(256) void k_cvt_w(const float* __restrict__ wqkv,
                                               const float* __restrict__ wout,
                                               _Float16* __restrict__ wqkv_h,
                                               _Float16* __restrict__ wout_h) {
  int i = blockIdx.x * 256 + threadIdx.x;
  if (i < QKV_M * CCH) wqkv_h[i] = (_Float16)wqkv[i];
  if (i < CCH * HIDDEN) wout_h[i] = (_Float16)wout[i];
}

// x[b][c][p] f32  ->  xt[b][p][c] f16
__global__ __launch_bounds__(256) void k_transpose_x(const float* __restrict__ x,
                                                     _Float16* __restrict__ xt) {
  size_t i = (size_t)blockIdx.x * 256 + threadIdx.x;   // over 16*512*1024
  int b = (int)(i >> 19);
  int r = (int)(i & ((1u << 19) - 1));
  int c = r >> 10;
  int p = r & 1023;
  xt[((size_t)b * NPIX + p) * CCH + c] = (_Float16)x[i];
}

// ---------------- Stage 1: QKV projection GEMM ----------------
// QKV[o][p] = sum_c Wqkv[o][c] * X[c][p]; stored transposed as QKVt[b][p][o]
// (f16), Q part pre-scaled by 1/sqrt(d); V part also scattered to
// Varr[b][c''][n]. A (weight) tile staged per k-step via the TDM into LDS.
__global__ __launch_bounds__(256) void k_gemm_qkv(const _Float16* __restrict__ W,
                                                  const _Float16* __restrict__ Xt,
                                                  _Float16* __restrict__ QKVt,
                                                  _Float16* __restrict__ Varr) {
  __shared__ _Float16 Asm[128 * 32];   // 8 KB staged weight tile
  const int b = blockIdx.z;
  const int tid = threadIdx.x;
  const int wave = tid >> 5, lane = tid & 31;
  const int lane_lo = lane & 15, lane_hi = lane >> 4;
  const int mblk = blockIdx.x * 128;
  const int m0l = (wave >> 1) * 32;               // wave's M offset inside block
  const int m0 = mblk + m0l;
  const int n0 = blockIdx.y * 64 + (wave & 1) * 32;
  const _Float16* Xb = Xt + (size_t)b * NPIX * CCH;
  const uint32_t lds_off = (uint32_t)(uintptr_t)&Asm[0];

  v8f acc[2][2] = {};
  for (int k0 = 0; k0 < CCH; k0 += 32) {
    if (wave == 0) {
      tdm_load_2d_f16(lds_off, W + (size_t)mblk * CCH + k0,
                      /*tensor_d0=*/32, /*tensor_d1=*/128,
                      /*tile_d0=*/32, /*tile_d1=*/128, /*stride0=*/CCH);
      __builtin_amdgcn_s_wait_tensorcnt(0);
    }
    __syncthreads();
    v16h a0 = frag_a(&Asm[(m0l + lane_lo) * 32], lane_hi);
    v16h a1 = frag_a(&Asm[(m0l + 16 + lane_lo) * 32], lane_hi);
    v16h b0 = frag_b(Xb + (size_t)(n0 + lane_lo) * CCH + k0, lane_hi);
    v16h b1 = frag_b(Xb + (size_t)(n0 + 16 + lane_lo) * CCH + k0, lane_hi);
    acc[0][0] = wmma16(a0, b0, acc[0][0]);
    acc[0][1] = wmma16(a0, b1, acc[0][1]);
    acc[1][0] = wmma16(a1, b0, acc[1][0]);
    acc[1][1] = wmma16(a1, b1, acc[1][1]);
    __syncthreads();
  }
#pragma unroll
  for (int mi = 0; mi < 2; ++mi)
#pragma unroll
    for (int ni = 0; ni < 2; ++ni) {
      int nn = n0 + ni * 16 + lane_lo;
#pragma unroll
      for (int r = 0; r < 8; ++r) {
        int mm = m0 + mi * 16 + r + lane_hi * 8;
        float v = acc[mi][ni][r];
        if (mm < HIDDEN) v *= ATTN_SCALE;   // fold softmax scale into Q
        _Float16 hv = (_Float16)v;
        QKVt[((size_t)b * NPIX + nn) * QKV_M + mm] = hv;
        if (mm >= 2 * HIDDEN)
          Varr[((size_t)b * HIDDEN + (mm - 2 * HIDDEN)) * NPIX + nn] = hv;
      }
    }
}

// ---------------- Stage 2: flash attention ----------------
// grid.x = 8 (i-blocks of 128), grid.y = 128 (b*8+h). Each wave owns 16 query rows.
__global__ __launch_bounds__(256) void k_attn(const _Float16* __restrict__ QKVt,
                                              const _Float16* __restrict__ Varr,
                                              _Float16* __restrict__ Aout) {
  __shared__ _Float16 pbuf[8][16][32];   // [wave][i][j]  8 KB
  const int bh = blockIdx.y;
  const int b = bh >> 3, h = bh & 7;
  const int tid = threadIdx.x;
  const int wave = tid >> 5, lane = tid & 31;
  const int lane_lo = lane & 15, lane_hi = lane >> 4;
  const int i0 = blockIdx.x * 128 + wave * 16;

  const _Float16* Qb = QKVt + (size_t)b * NPIX * QKV_M + h * HEAD_DIM;            // + i*QKV_M
  const _Float16* Kb = QKVt + (size_t)b * NPIX * QKV_M + HIDDEN + h * HEAD_DIM;   // + j*QKV_M
  const _Float16* Vb = Varr + ((size_t)b * HIDDEN + h * HEAD_DIM) * NPIX;         // + d*NPIX

  // loop-invariant Q as B-operand (col i, K=d)
  v16h qb0 = frag_b(Qb + (size_t)(i0 + lane_lo) * QKV_M + 0, lane_hi);
  v16h qb1 = frag_b(Qb + (size_t)(i0 + lane_lo) * QKV_M + 32, lane_hi);

  v8f oacc[4] = {};                 // O^T tiles: d rows 0..63, col i
  float mrun = -1e30f, lrun = 0.0f;

  for (int j0 = 0; j0 < NPIX; j0 += 32) {
    // S^T[j, i] for 32 j-rows: two 16x16 tiles, K=64 split into two WMMAs
    v8f s[2];
#pragma unroll
    for (int t = 0; t < 2; ++t) {
      v16h ka0 = frag_a(Kb + (size_t)(j0 + 16 * t + lane_lo) * QKV_M + 0, lane_hi);
      v16h ka1 = frag_a(Kb + (size_t)(j0 + 16 * t + lane_lo) * QKV_M + 32, lane_hi);
      v8f z = {};
      z = wmma16(ka0, qb0, z);
      z = wmma16(ka1, qb1, z);
      s[t] = z;
    }
    // per-lane online softmax over the 16 j-values this lane holds for its column i
    float mloc = mrun;
#pragma unroll
    for (int t = 0; t < 2; ++t)
#pragma unroll
      for (int r = 0; r < 8; ++r) mloc = fmaxf(mloc, s[t][r]);
    float corr = __expf(mrun - mloc);
    mrun = mloc;
    lrun *= corr;
#pragma unroll
    for (int dt = 0; dt < 4; ++dt)
#pragma unroll
      for (int r = 0; r < 8; ++r) oacc[dt][r] *= corr;
    // P -> LDS (per-wave buffer, j-contiguous so it reloads as a B fragment)
#pragma unroll
    for (int t = 0; t < 2; ++t)
#pragma unroll
      for (int r = 0; r < 8; ++r) {
        float pv = __expf(s[t][r] - mrun);
        lrun += pv;
        int jl = 16 * t + r + lane_hi * 8;
        pbuf[wave][lane_lo][jl] = (_Float16)pv;
      }
    __syncthreads();   // uniform; orders LDS writes vs cross-lane reads
    v16h pb = frag_b(&pbuf[wave][lane_lo][0], lane_hi);   // B: col i, K=j(32)
#pragma unroll
    for (int dt = 0; dt < 4; ++dt) {
      v16h va = frag_a(Vb + (size_t)(dt * 16 + lane_lo) * NPIX + j0, lane_hi);
      oacc[dt] = wmma16(va, pb, oacc[dt]);
    }
    __syncthreads();
  }

  float inv = 1.0f / lrun;
  const int ig = i0 + lane_lo;
  _Float16* dst = Aout + ((size_t)b * NPIX + ig) * HIDDEN + h * HEAD_DIM;
#pragma unroll
  for (int dt = 0; dt < 4; ++dt)
#pragma unroll
    for (int r = 0; r < 8; ++r) {
      int d = dt * 16 + r + lane_hi * 8;
      dst[d] = (_Float16)(oacc[dt][r] * inv);
    }
}

// ---------------- Stage 3: output projection GEMM (+bias) ----------------
__global__ __launch_bounds__(256) void k_gemm_out(const _Float16* __restrict__ Wo,
                                                  const _Float16* __restrict__ At,
                                                  const float* __restrict__ bias,
                                                  float* __restrict__ Y) {
  __shared__ _Float16 Asm[128 * 32];   // 8 KB staged weight tile
  const int b = blockIdx.z;
  const int tid = threadIdx.x;
  const int wave = tid >> 5, lane = tid & 31;
  const int lane_lo = lane & 15, lane_hi = lane >> 4;
  const int mblk = blockIdx.x * 128;
  const int m0l = (wave >> 1) * 32;
  const int m0 = mblk + m0l;
  const int n0 = blockIdx.y * 64 + (wave & 1) * 32;
  const _Float16* Ab = At + (size_t)b * NPIX * HIDDEN;
  const uint32_t lds_off = (uint32_t)(uintptr_t)&Asm[0];

  v8f acc[2][2] = {};
  for (int k0 = 0; k0 < HIDDEN; k0 += 32) {
    if (wave == 0) {
      tdm_load_2d_f16(lds_off, Wo + (size_t)mblk * HIDDEN + k0,
                      32, 128, 32, 128, HIDDEN);
      __builtin_amdgcn_s_wait_tensorcnt(0);
    }
    __syncthreads();
    v16h a0 = frag_a(&Asm[(m0l + lane_lo) * 32], lane_hi);
    v16h a1 = frag_a(&Asm[(m0l + 16 + lane_lo) * 32], lane_hi);
    v16h b0 = frag_b(Ab + (size_t)(n0 + lane_lo) * HIDDEN + k0, lane_hi);
    v16h b1 = frag_b(Ab + (size_t)(n0 + 16 + lane_lo) * HIDDEN + k0, lane_hi);
    acc[0][0] = wmma16(a0, b0, acc[0][0]);
    acc[0][1] = wmma16(a0, b1, acc[0][1]);
    acc[1][0] = wmma16(a1, b0, acc[1][0]);
    acc[1][1] = wmma16(a1, b1, acc[1][1]);
    __syncthreads();
  }
#pragma unroll
  for (int mi = 0; mi < 2; ++mi)
#pragma unroll
    for (int ni = 0; ni < 2; ++ni) {
      int nn = n0 + ni * 16 + lane_lo;
#pragma unroll
      for (int r = 0; r < 8; ++r) {
        int mm = m0 + mi * 16 + r + lane_hi * 8;
        Y[((size_t)b * CCH + mm) * NPIX + nn] = acc[mi][ni][r] + bias[mm];
      }
    }
}

// ---------------- Stage 4: RMSNorm (eps = ATTN_SCALE, faithful) + residual ----------------
__global__ __launch_bounds__(256) void k_rmsnorm(float* __restrict__ Y,
                                                 const float* __restrict__ x,
                                                 const float* __restrict__ g) {
  __shared__ float red[256];
  const int bp = blockIdx.x;              // b*1024 + p
  const int b = bp >> 10, p = bp & 1023;
  const float* xcol = x + (size_t)b * CCH * NPIX + p;
  float* ycol = Y + (size_t)b * CCH * NPIX + p;
  const int tid = threadIdx.x;
  float v0 = ycol[(size_t)tid * NPIX];
  float v1 = ycol[(size_t)(tid + 256) * NPIX];
  red[tid] = v0 * v0 + v1 * v1;
  __syncthreads();
  for (int s = 128; s > 0; s >>= 1) {
    if (tid < s) red[tid] += red[tid + s];
    __syncthreads();
  }
  float inv = rsqrtf(red[0] * (1.0f / CCH) + ATTN_SCALE);
  ycol[(size_t)tid * NPIX] = v0 * inv * g[tid] + xcol[(size_t)tid * NPIX];
  ycol[(size_t)(tid + 256) * NPIX] = v1 * inv * g[tid + 256] + xcol[(size_t)(tid + 256) * NPIX];
}

// ---------------- host launch ----------------

extern "C" void kernel_launch(void* const* d_in, const int* in_sizes, int n_in,
                              void* d_out, int out_size, void* d_ws, size_t ws_size,
                              hipStream_t stream) {
  (void)in_sizes; (void)n_in; (void)out_size; (void)ws_size;
  const float* x      = (const float*)d_in[0];   // [16,512,32,32]
  const float* w_qkv  = (const float*)d_in[1];   // [1536,512]
  const float* w_out  = (const float*)d_in[2];   // [512,512]
  const float* b_out  = (const float*)d_in[3];   // [512]
  const float* rms_g  = (const float*)d_in[4];   // [512]
  float* y = (float*)d_out;                      // [16,512,32,32]

  char* ws = (char*)d_ws;
  size_t off = 0;
  auto take = [&](size_t bytes) {
    char* p = ws + off;
    off = (off + bytes + 255) & ~(size_t)255;
    return p;
  };
  _Float16* wqkv_h = (_Float16*)take((size_t)QKV_M * CCH * 2);
  _Float16* wout_h = (_Float16*)take((size_t)CCH * HIDDEN * 2);
  _Float16* xt     = (_Float16*)take((size_t)NB * NPIX * CCH * 2);
  _Float16* qkvt   = (_Float16*)take((size_t)NB * NPIX * QKV_M * 2);
  _Float16* varr   = (_Float16*)take((size_t)NB * HIDDEN * NPIX * 2);
  _Float16* aout   = (_Float16*)take((size_t)NB * NPIX * HIDDEN * 2);

  k_cvt_w<<<(QKV_M * CCH) / 256, 256, 0, stream>>>(w_qkv, w_out, wqkv_h, wout_h);
  k_transpose_x<<<(NB * CCH * NPIX) / 256, 256, 0, stream>>>(x, xt);
  k_gemm_qkv<<<dim3(QKV_M / 128, NPIX / 64, NB), 256, 0, stream>>>(wqkv_h, xt, qkvt, varr);
  k_attn<<<dim3(NPIX / 128, NB * HEADS), 256, 0, stream>>>(qkvt, varr, aout);
  k_gemm_out<<<dim3(CCH / 128, NPIX / 64, NB), 256, 0, stream>>>(wout_h, aout, b_out, y);
  k_rmsnorm<<<NB * NPIX, 256, 0, stream>>>(y, x, rms_g);
}
